// RWKVAttention_37151467110715
// MI455X (gfx1250) — compile-verified, matrix operand order checked by reading, not run on hardware
//
#include <hip/hip_runtime.h>
#include <hip/hip_bf16.h>

typedef __bf16 bf16_t;
typedef __bf16 v16bf __attribute__((ext_vector_type(16)));
typedef __bf16 v8bf  __attribute__((ext_vector_type(8)));
typedef float  v8f   __attribute__((ext_vector_type(8)));

#define C_DIM 2048
#define B_DIM 4
#define T_DIM 2048
#define M_DIM (B_DIM * T_DIM)   // 8192 rows

// GEMM tiling
#define BM 128
#define BN 128
#define BK 32
#define LDK (BK + 8)            // LDS row stride (bf16 elems), keeps 16B alignment, avoids bank conflicts

// WKV chunked scan
#define NCH 16
#define CHL (T_DIM / NCH)       // 128 steps per chunk

// ---------------------------------------------------------------------------
// CDNA5 async global->LDS copy (ASYNCcnt path), 16 bytes per lane.
// GVS addressing: mem = SGPR64 base + VGPR32 byte offset.
// LDS address operand = low 32 bits of the generic pointer (ISA: flat LDS
// addresses carry the LDS offset in addr[31:0]).
// ---------------------------------------------------------------------------
__device__ __forceinline__ void async_copy_b128(const void* gbase, unsigned goff,
                                                void* lptr) {
    unsigned loff = (unsigned)(uintptr_t)lptr;
    asm volatile("global_load_async_to_lds_b128 %0, %1, %2"
                 :
                 : "v"(loff), "v"(goff), "s"(gbase)
                 : "memory");
}

__device__ __forceinline__ void wait_asynccnt0() {
    asm volatile("s_wait_asynccnt 0x0" ::: "memory");
}

// ---------------------------------------------------------------------------
// fp32 -> bf16 conversion (weights)
// ---------------------------------------------------------------------------
__global__ __launch_bounds__(256)
void f32_to_bf16_kernel(const float* __restrict__ in, bf16_t* __restrict__ out, int n) {
    int i = (blockIdx.x * 256 + threadIdx.x) * 4;
    if (i + 3 < n) {
        float4 v = *(const float4*)(in + i);
        out[i + 0] = (bf16_t)v.x;
        out[i + 1] = (bf16_t)v.y;
        out[i + 2] = (bf16_t)v.z;
        out[i + 3] = (bf16_t)v.w;
    }
}

// ---------------------------------------------------------------------------
// Tiled bf16 WMMA GEMM:  Out[m,n] = sum_k A[m,k] * W[n,k]   (x @ W.T)
// MIX=true : A built on the fly from fp32 x via time-shift mix, cast to bf16
// MIX=false: A is an existing bf16 [M,C] buffer (async-copied straight to LDS)
// B tile (bf16 weights) is always async-copied straight to LDS.
// ---------------------------------------------------------------------------
template <bool MIX>
__global__ __launch_bounds__(256)
void gemm_bf16_wmma(const float* __restrict__ X,      // [M,C] fp32 (MIX)
                    const bf16_t* __restrict__ Abf,   // [M,C] bf16 (!MIX)
                    const float* __restrict__ tmix,   // [C]   (MIX)
                    const bf16_t* __restrict__ W,     // [C,C] bf16, row-major [n][k]
                    float* __restrict__ Out)          // [M,C] fp32
{
    __shared__ __align__(16) bf16_t sA[2][BM][LDK];
    __shared__ __align__(16) bf16_t sB[2][BN][LDK];

    const int tid  = threadIdx.x;
    const int lane = tid & 31;
    const int wave = tid >> 5;
    const int wm   = wave >> 2;        // 0..1  -> wave M block of 64
    const int wn   = wave & 3;         // 0..3  -> wave N block of 32
    const int bm   = blockIdx.y * BM;
    const int bn   = blockIdx.x * BN;

    // staging: each thread owns one 16-wide half row of A tile and of B tile
    const int srow = tid >> 1;         // 0..127
    const int scol = (tid & 1) * 16;   // 0 or 16

    const int grow = bm + srow;                       // global A row
    const int t    = grow & (T_DIM - 1);
    const int prow = (t == 0) ? (grow + 1) : (grow - 1);  // time-shift source row
    const int gbn  = bn + srow;                       // global W row (N index)

    v8f acc[4][2];
#pragma unroll
    for (int i = 0; i < 4; ++i)
#pragma unroll
        for (int j = 0; j < 2; ++j)
            acc[i][j] = v8f{0.f, 0.f, 0.f, 0.f, 0.f, 0.f, 0.f, 0.f};

    auto loadTile = [&](int kt, int buf) {
        const int gk = kt * BK + scol;
        // ---- A tile ----
        if (MIX) {
            const float4* xp = (const float4*)(X + (size_t)grow * C_DIM + gk);
            const float4* pp = (const float4*)(X + (size_t)prow * C_DIM + gk);
            const float4* tp = (const float4*)(tmix + gk);
            __builtin_prefetch(X + (size_t)grow * C_DIM + gk + 2 * BK, 0, 1);
            v8bf lo, hi;
#pragma unroll
            for (int q = 0; q < 2; ++q) {
                float4 xv = xp[q], pv = pp[q], tv = tp[q];
                lo[q * 4 + 0] = (bf16_t)(xv.x + (pv.x - xv.x) * tv.x);
                lo[q * 4 + 1] = (bf16_t)(xv.y + (pv.y - xv.y) * tv.y);
                lo[q * 4 + 2] = (bf16_t)(xv.z + (pv.z - xv.z) * tv.z);
                lo[q * 4 + 3] = (bf16_t)(xv.w + (pv.w - xv.w) * tv.w);
            }
#pragma unroll
            for (int q = 0; q < 2; ++q) {
                float4 xv = xp[2 + q], pv = pp[2 + q], tv = tp[2 + q];
                hi[q * 4 + 0] = (bf16_t)(xv.x + (pv.x - xv.x) * tv.x);
                hi[q * 4 + 1] = (bf16_t)(xv.y + (pv.y - xv.y) * tv.y);
                hi[q * 4 + 2] = (bf16_t)(xv.z + (pv.z - xv.z) * tv.z);
                hi[q * 4 + 3] = (bf16_t)(xv.w + (pv.w - xv.w) * tv.w);
            }
            *(v8bf*)&sA[buf][srow][scol + 0] = lo;
            *(v8bf*)&sA[buf][srow][scol + 8] = hi;
        } else {
            const unsigned gofA =
                (unsigned)(((size_t)grow * C_DIM + gk) * sizeof(bf16_t));
            async_copy_b128(Abf, gofA + 0,  &sA[buf][srow][scol + 0]);
            async_copy_b128(Abf, gofA + 16, &sA[buf][srow][scol + 8]);
            __builtin_prefetch(Abf + (size_t)grow * C_DIM + gk + 2 * BK, 0, 1);
        }
        // ---- B tile (weights, already bf16, [n][k]) -> async straight to LDS ----
        const unsigned gofB = (unsigned)(((size_t)gbn * C_DIM + gk) * sizeof(bf16_t));
        async_copy_b128(W, gofB + 0,  &sB[buf][srow][scol + 0]);
        async_copy_b128(W, gofB + 16, &sB[buf][srow][scol + 8]);
        __builtin_prefetch(W + (size_t)gbn * C_DIM + gk + 2 * BK, 0, 1);
    };

    // fragment gather coords (16-bit A-matrix 16x32 layout: lanes 0-15 K={0..7,16..23},
    // lanes 16-31 K={8..15,24..31})
    const int fr = lane & 15;
    const int fk = (lane >> 4) * 8;

    auto compute = [&](int buf) {
        v16bf afrag[4];
        v16bf bfrag[2];
#pragma unroll
        for (int i = 0; i < 4; ++i) {
            const bf16_t* pa = &sA[buf][wm * 64 + i * 16 + fr][fk];
            v8bf lo = *(const v8bf*)pa;
            v8bf hi = *(const v8bf*)(pa + 16);
#pragma unroll
            for (int e = 0; e < 8; ++e) { afrag[i][e] = lo[e]; afrag[i][e + 8] = hi[e]; }
        }
#pragma unroll
        for (int j = 0; j < 2; ++j) {
            const bf16_t* pb = &sB[buf][wn * 32 + j * 16 + fr][fk];
            v8bf lo = *(const v8bf*)pb;
            v8bf hi = *(const v8bf*)(pb + 16);
#pragma unroll
            for (int e = 0; e < 8; ++e) { bfrag[j][e] = lo[e]; bfrag[j][e + 8] = hi[e]; }
        }
#pragma unroll
        for (int i = 0; i < 4; ++i)
#pragma unroll
            for (int j = 0; j < 2; ++j)
                acc[i][j] = __builtin_amdgcn_wmma_f32_16x16x32_bf16(
                    false, afrag[i], false, bfrag[j], (short)0, acc[i][j], false, false);
    };

    const int nk = C_DIM / BK;   // 64 K-steps
    loadTile(0, 0);
    for (int kt = 0; kt < nk; ++kt) {
        wait_asynccnt0();     // async writes to the upcoming buffer are done
        __syncthreads();
        if (kt + 1 < nk) loadTile(kt + 1, (kt + 1) & 1);
        compute(kt & 1);
    }

    // epilogue: C/D layout -> element (vgpr r, lane l): M = (l>>4)*8 + r, N = (l&15)
    const int en = lane & 15;
    const int em = (lane >> 4) * 8;
#pragma unroll
    for (int i = 0; i < 4; ++i) {
#pragma unroll
        for (int j = 0; j < 2; ++j) {
            const int mb = bm + wm * 64 + i * 16 + em;
            const int nb = bn + wn * 32 + j * 16 + en;
#pragma unroll
            for (int r = 0; r < 8; ++r)
                Out[(size_t)(mb + r) * C_DIM + nb] = acc[i][j][r];
        }
    }
}

// ---------------------------------------------------------------------------
// WKV chunked scan (3 passes). Recurrence: a' = w*a + e^k*v, b' = w*b + e^k.
// ---------------------------------------------------------------------------
__device__ __forceinline__ float sigmoidf_fast(float x) {
    return 1.f / (1.f + __expf(-x));
}

// Pass 1: per-(b,chunk,c) local decayed sums. idx = (b*NCH + ch)*C + c
__global__ __launch_bounds__(256)
void wkv_pass1(const float* __restrict__ K, const float* __restrict__ V,
               const float* __restrict__ td,
               float* __restrict__ Sa, float* __restrict__ Sb) {
    const int idx = blockIdx.x * 256 + threadIdx.x;
    const int c   = idx % C_DIM;
    const int ch  = (idx / C_DIM) % NCH;
    const int b   = idx / (C_DIM * NCH);
    const float w = __expf(-__expf(td[c]));
    float a = 0.f, s = 0.f;
    size_t off = ((size_t)b * T_DIM + (size_t)ch * CHL) * C_DIM + c;
    for (int i = 0; i < CHL; ++i, off += C_DIM) {
        const float ek = __expf(K[off]);
        a = w * a + ek * V[off];
        s = w * s + ek;
    }
    Sa[idx] = a;
    Sb[idx] = s;
}

// Pass 2: per-(b,c) sequential combine over the 16 chunks -> chunk-start states
__global__ __launch_bounds__(256)
void wkv_pass2(const float* __restrict__ td,
               const float* __restrict__ Sa, const float* __restrict__ Sb,
               float* __restrict__ A0, float* __restrict__ B0) {
    const int idx = blockIdx.x * 256 + threadIdx.x;  // b*C + c
    const int c   = idx % C_DIM;
    const int b   = idx / C_DIM;
    const float w  = __expf(-__expf(td[c]));
    const float wc = __powf(w, (float)CHL);
    float a = 0.f, s = 0.f;
    for (int ch = 0; ch < NCH; ++ch) {
        const int j = (b * NCH + ch) * C_DIM + c;
        A0[j] = a;
        B0[j] = s;
        a = wc * a + Sa[j];
        s = wc * s + Sb[j];
    }
}

// Pass 3: replay each chunk from its start state, fuse sigmoids + gate,
// emit attn as bf16 (A operand of the output GEMM).
__global__ __launch_bounds__(256)
void wkv_pass3(const float* __restrict__ R, const float* __restrict__ K,
               const float* __restrict__ V, const float* __restrict__ G,
               const float* __restrict__ td, const float* __restrict__ tf,
               const float* __restrict__ A0, const float* __restrict__ B0,
               bf16_t* __restrict__ Attn) {
    const int idx = blockIdx.x * 256 + threadIdx.x;
    const int c   = idx % C_DIM;
    const int ch  = (idx / C_DIM) % NCH;
    const int b   = idx / (C_DIM * NCH);
    const float w  = __expf(-__expf(td[c]));
    const float eu = __expf(tf[c]);
    float a = A0[idx], s = B0[idx];
    size_t off = ((size_t)b * T_DIM + (size_t)ch * CHL) * C_DIM + c;
    for (int i = 0; i < CHL; ++i, off += C_DIM) {
        const float kt = K[off], vt = V[off];
        const float ek = __expf(kt);
        const float euek = eu * ek;
        const float wkv = (a + euek * vt) / (s + euek);
        const float sr = sigmoidf_fast(R[off]);
        const float sg = sigmoidf_fast(G[off]);
        Attn[off] = (bf16_t)(sr * wkv * sg);
        a = w * a + ek * vt;
        s = w * s + ek;
    }
}

// ---------------------------------------------------------------------------
// Row LayerNorm: one block per row of o[M,C]
// ---------------------------------------------------------------------------
__global__ __launch_bounds__(256)
void layernorm_kernel(const float* __restrict__ O,
                      const float* __restrict__ gamma, const float* __restrict__ beta,
                      float* __restrict__ Out) {
    __shared__ float redS[8];
    __shared__ float redQ[8];
    const int row = blockIdx.x;
    const float* o = O + (size_t)row * C_DIM;
    float s = 0.f, q = 0.f;
    for (int i = threadIdx.x; i < C_DIM; i += 256) {
        const float v = o[i];
        s += v;
        q += v * v;
    }
#pragma unroll
    for (int off = 16; off > 0; off >>= 1) {
        s += __shfl_down(s, off, 32);
        q += __shfl_down(q, off, 32);
    }
    const int lane = threadIdx.x & 31, wave = threadIdx.x >> 5;
    if (lane == 0) { redS[wave] = s; redQ[wave] = q; }
    __syncthreads();
    if (wave == 0) {
        s = (lane < 8) ? redS[lane] : 0.f;
        q = (lane < 8) ? redQ[lane] : 0.f;
#pragma unroll
        for (int off = 4; off > 0; off >>= 1) {
            s += __shfl_down(s, off, 32);
            q += __shfl_down(q, off, 32);
        }
        if (lane == 0) { redS[0] = s; redQ[0] = q; }
    }
    __syncthreads();
    const float mu  = redS[0] * (1.f / C_DIM);
    const float var = redQ[0] * (1.f / C_DIM) - mu * mu;
    const float rs  = rsqrtf(var + 1e-5f);
    for (int i = threadIdx.x; i < C_DIM; i += 256)
        Out[(size_t)row * C_DIM + i] = (o[i] - mu) * rs * gamma[i] + beta[i];
}

// ---------------------------------------------------------------------------
// Launch
// ---------------------------------------------------------------------------
extern "C" void kernel_launch(void* const* d_in, const int* in_sizes, int n_in,
                              void* d_out, int out_size, void* d_ws, size_t ws_size,
                              hipStream_t stream) {
    const float* x          = (const float*)d_in[0];
    const float* time_decay = (const float*)d_in[1];
    const float* time_first = (const float*)d_in[2];
    const float* tm_k       = (const float*)d_in[3];
    const float* tm_v       = (const float*)d_in[4];
    const float* tm_r       = (const float*)d_in[5];
    const float* tm_g       = (const float*)d_in[6];
    const float* Wsrc[5]    = {(const float*)d_in[7], (const float*)d_in[8],
                               (const float*)d_in[9], (const float*)d_in[10],
                               (const float*)d_in[11]};  // Wr, Wk, Wv, Wg, Wo
    const float* ln_g       = (const float*)d_in[12];
    const float* ln_b       = (const float*)d_in[13];
    float* out              = (float*)d_out;

    const size_t CC = (size_t)C_DIM * C_DIM;   // 4.19M
    const size_t MC = (size_t)M_DIM * C_DIM;   // 16.8M

    // workspace layout
    char* ws = (char*)d_ws;
    bf16_t* wbf0 = (bf16_t*)ws;                               // 5*CC bf16 (40MB)
    float*  Rb   = (float*)(ws + 5 * CC * sizeof(bf16_t));    // 4*MC f32 (256MB)
    float*  Kb   = Rb + MC;
    float*  Vb   = Kb + MC;
    float*  Gb   = Vb + MC;
    bf16_t* attn = (bf16_t*)(Gb + MC);                        // MC bf16 (32MB)
    float*  Sa   = (float*)(attn + MC);                       // 4 * B*C*NCH f32 (2MB)
    float*  Sb   = Sa + (size_t)B_DIM * C_DIM * NCH;
    float*  A0   = Sb + (size_t)B_DIM * C_DIM * NCH;
    float*  B0   = A0 + (size_t)B_DIM * C_DIM * NCH;
    float*  Ob   = Rb;  // output-GEMM result reuses R buffer

    // 1) weights fp32 -> bf16 (reused across K=2048 reduction; L2-resident)
    for (int i = 0; i < 5; ++i)
        f32_to_bf16_kernel<<<(unsigned)(CC / 1024), 256, 0, stream>>>(
            Wsrc[i], wbf0 + (size_t)i * CC, (int)CC);

    // 2) four input GEMMs with fused time-shift mix
    dim3 ggrid(C_DIM / BN, M_DIM / BM);  // (16, 64)
    gemm_bf16_wmma<true><<<ggrid, 256, 0, stream>>>(x, nullptr, tm_r, wbf0 + 0 * CC, Rb);
    gemm_bf16_wmma<true><<<ggrid, 256, 0, stream>>>(x, nullptr, tm_k, wbf0 + 1 * CC, Kb);
    gemm_bf16_wmma<true><<<ggrid, 256, 0, stream>>>(x, nullptr, tm_v, wbf0 + 2 * CC, Vb);
    gemm_bf16_wmma<true><<<ggrid, 256, 0, stream>>>(x, nullptr, tm_g, wbf0 + 3 * CC, Gb);

    // 3) chunk-parallel WKV scan fused with sigmoids + gating -> attn (bf16)
    const unsigned p13 = (unsigned)((size_t)B_DIM * NCH * C_DIM / 256);  // 512 blocks
    const unsigned p2  = (unsigned)((size_t)B_DIM * C_DIM / 256);        // 32 blocks
    wkv_pass1<<<p13, 256, 0, stream>>>(Kb, Vb, time_decay, Sa, Sb);
    wkv_pass2<<<p2, 256, 0, stream>>>(time_decay, Sa, Sb, A0, B0);
    wkv_pass3<<<p13, 256, 0, stream>>>(Rb, Kb, Vb, Gb, time_decay, time_first,
                                       A0, B0, attn);

    // 4) output GEMM (bf16 A path) + LayerNorm
    gemm_bf16_wmma<false><<<ggrid, 256, 0, stream>>>(nullptr, attn, nullptr,
                                                     wbf0 + 4 * CC, Ob);
    layernorm_kernel<<<M_DIM, 256, 0, stream>>>(Ob, ln_g, ln_b, out);
}
